// EnhancedLPMP_50053548867804
// MI455X (gfx1250) — compile-verified, matrix operand order checked by reading, not run on hardware
//
#include <hip/hip_runtime.h>
#include <hip/hip_bf16.h>
#include <math.h>

typedef __attribute__((ext_vector_type(16))) _Float16 v16h;
typedef __attribute__((ext_vector_type(8)))  _Float16 v8h;
typedef __attribute__((ext_vector_type(2)))  _Float16 v2h;
typedef __attribute__((ext_vector_type(8)))  float    v8f;

#define BB   256
#define HH   32
#define WW   32
#define HW   1024
#define TN   18
#define KPAD 32
#define NB   8
#define NF   64
#define FD   256

__device__ __forceinline__ float sigmoidf_(float x) { return 1.0f / (1.0f + __expf(-x)); }
__device__ __forceinline__ float softplusf_(float x) {
  return (x > 20.0f) ? x : log1pf(__expf(x));
}

// ---------------------------------------------------------------------------
// Kernel A: central-difference gradient magnitude + InstanceNorm per (b,c)
// ---------------------------------------------------------------------------
__global__ __launch_bounds__(256) void gradnorm_kernel(
    const float* __restrict__ x,
    const float* __restrict__ norm_w,
    const float* __restrict__ norm_b,
    float* __restrict__ gmap)
{
  __shared__ float xi[HW];
  __shared__ float s_sum, s_sqs;
  const int img = blockIdx.x;          // b*3 + c
  const int c   = img % 3;
  const int tid = threadIdx.x;
  const float* xin = x + (size_t)img * HW;

  for (int i = tid; i < HW; i += 256) xi[i] = xin[i];
  if (tid == 0) { s_sum = 0.0f; s_sqs = 0.0f; }
  __syncthreads();

  float lsum = 0.0f, lsqs = 0.0f;
  float vals[4];
#pragma unroll
  for (int k = 0; k < 4; ++k) {
    int p = tid + k * 256;
    int y = p >> 5, xx = p & 31;
    float gx = ((xx < 31) ? xi[y * 32 + xx + 1] : 0.0f) -
               ((xx > 0)  ? xi[y * 32 + xx - 1] : 0.0f);
    float gy = ((y < 31) ? xi[(y + 1) * 32 + xx] : 0.0f) -
               ((y > 0)  ? xi[(y - 1) * 32 + xx] : 0.0f);
    float m = sqrtf(gx * gx + gy * gy + 1e-8f);
    vals[k] = m; lsum += m; lsqs += m * m;
  }
  for (int mk = 1; mk < 32; mk <<= 1) {
    lsum += __shfl_xor(lsum, mk, 32);
    lsqs += __shfl_xor(lsqs, mk, 32);
  }
  if ((tid & 31) == 0) { atomicAdd(&s_sum, lsum); atomicAdd(&s_sqs, lsqs); }
  __syncthreads();

  float mu   = s_sum * (1.0f / HW);
  float var  = s_sqs * (1.0f / HW) - mu * mu;
  float rstd = rsqrtf(var + 1e-5f);
  float nw = norm_w[c], nb = norm_b[c];
  float* gout = gmap + (size_t)img * HW;
#pragma unroll
  for (int k = 0; k < 4; ++k) {
    int p = tid + k * 256;
    gout[p] = (vals[k] - mu) * rstd * nw + nb;
  }
}

// ---------------------------------------------------------------------------
// Kernel B: per-image fused pipeline. soft_q is stored split so that WMMA
// A-fragments are two unconditional ds_load_b128 per lane (row-major 16-bit
// layout: lane<16 -> K{0..7,16..23} of row lane; lane>=16 -> K{8..15,24..31}).
//   piece1[p][16] : K 0..15
//   piece2[p][8]  : K 16..23 (18..23 zero); row 1024 = all-zero row used by
//                   lanes>=16 whose second chunk is K 24..31 (pure padding).
// ---------------------------------------------------------------------------
__global__ __launch_bounds__(512) void lpmp_main_kernel(
    const float* __restrict__ gmap,
    const float* __restrict__ radius_param,
    const float* __restrict__ bin_size_param,
    const float* __restrict__ soft_temp_param,
    const float* __restrict__ bin_centers,
    const float* __restrict__ offset_w,
    const float* __restrict__ offset_b,
    const float* __restrict__ se_w1,
    const float* __restrict__ se_w2,
    const float* __restrict__ agg_w1,
    const float* __restrict__ agg_b1,
    const float* __restrict__ agg_w2,
    const float* __restrict__ agg_b2,
    const float* __restrict__ gem_p,
    float* __restrict__ gpool)
{
  __shared__ float    img[34 * 34];                      // zero-padded grad_map
  __shared__ __align__(16) _Float16 piece1[HW * 16];     // K 0..15
  __shared__ __align__(16) _Float16 piece2[(HW + 1) * 8];// K 16..23 (+ zero row)
  __shared__ __align__(16) _Float16 wB[NF * KPAD];       // agg_w1, K-padded to 32
  __shared__ float    cw[36 * 9];
  __shared__ float    cb[36];
  __shared__ float    b1s[NF];
  __shared__ float    w2s[NF];
  __shared__ float    red[TN];
  __shared__ float    sse[TN];
  __shared__ float    bc[NB];
  __shared__ float    pv[HW];
  __shared__ float    gacc;

  const int idx  = blockIdx.x;               // b*3 + c
  const int c    = idx % 3;
  const int tid  = threadIdx.x;
  const int lane = tid & 31;
  const int wav  = tid >> 5;
  const float* gin = gmap + (size_t)idx * HW;

  for (int i = tid; i < 34 * 34; i += 512) {
    int yy = i / 34 - 1, xx = i % 34 - 1;
    img[i] = (yy >= 0 && yy < 32 && xx >= 0 && xx < 32) ? gin[yy * 32 + xx] : 0.0f;
  }
  for (int i = tid; i < 36 * 9; i += 512) cw[i] = offset_w[(c * 36) * 9 + i];
  if (tid < 36) cb[tid] = offset_b[c * 36 + tid];
  if (tid < TN) red[tid] = 0.0f;
  if (tid < NB) bc[tid] = bin_centers[tid];
  if (tid < NF) { b1s[tid] = agg_b1[tid]; w2s[tid] = agg_w2[tid]; }
  for (int i = tid; i < NF * KPAD; i += 512) {
    int n = i >> 5, k = i & 31;
    wB[i] = (_Float16)((k < TN) ? agg_w1[n * TN + k] : 0.0f);
  }
  for (int i = tid; i < (HW + 1) * 8; i += 512) piece2[i] = (_Float16)0.0f;
  if (tid == 0) gacc = 0.0f;
  __syncthreads();

  // radii; grid-normalization constants cancel: pixel offset = tanh*4*radius
  const float amp0 = 4.0f * (0.5f + 4.5f * sigmoidf_(radius_param[0]));
  const float amp1 = 4.0f * (0.5f + 4.5f * sigmoidf_(radius_param[1]));

  float sacc[TN];
#pragma unroll
  for (int t = 0; t < TN; ++t) sacc[t] = 0.0f;

  for (int pp = 0; pp < 2; ++pp) {
    int p = tid + pp * 512;
    int y = p >> 5, xx = p & 31;
    float nb9[9];
#pragma unroll
    for (int j = 0; j < 9; ++j) nb9[j] = img[(y + j / 3) * 34 + (xx + j % 3)];
    _Float16 h16[TN];
#pragma unroll
    for (int t = 0; t < TN; ++t) {
      const float* wx = &cw[(2 * t) * 9];
      const float* wy = &cw[(2 * t + 1) * 9];
      float ox = cb[2 * t], oy = cb[2 * t + 1];
#pragma unroll
      for (int j = 0; j < 9; ++j) { ox += nb9[j] * wx[j]; oy += nb9[j] * wy[j]; }
      float amp = (t < 9) ? amp0 : amp1;
      float ix = fminf(fmaxf((float)xx + tanhf(ox) * amp, 0.0f), 31.0f);
      float iy = fminf(fmaxf((float)y  + tanhf(oy) * amp, 0.0f), 31.0f);
      int x0 = (int)ix, y0 = (int)iy;
      float fx = ix - (float)x0, fy = iy - (float)y0;
      int x1 = min(x0 + 1, 31), y1 = min(y0 + 1, 31);
      float v00 = img[(y0 + 1) * 34 + x0 + 1];
      float v01 = img[(y0 + 1) * 34 + x1 + 1];
      float v10 = img[(y1 + 1) * 34 + x0 + 1];
      float v11 = img[(y1 + 1) * 34 + x1 + 1];
      float sv = (v00 * (1.0f - fx) + v01 * fx) * (1.0f - fy) +
                 (v10 * (1.0f - fx) + v11 * fx) * fy;
      h16[t] = (_Float16)sv;
      sacc[t] += sv;
    }
    // vectorized row store: 2x b128 + 1x b32
    v8h r0, r1;
#pragma unroll
    for (int j = 0; j < 8; ++j) { r0[j] = h16[j]; r1[j] = h16[8 + j]; }
    *(v8h*)(piece1 + p * 16)     = r0;
    *(v8h*)(piece1 + p * 16 + 8) = r1;
    v2h r2; r2[0] = h16[16]; r2[1] = h16[17];
    *(v2h*)(piece2 + p * 8) = r2;
  }
  // SE mean reduction: shfl within wave, ds_add_f32 across waves
#pragma unroll
  for (int t = 0; t < TN; ++t) {
    float v = sacc[t];
    for (int mk = 1; mk < 32; mk <<= 1) v += __shfl_xor(v, mk, 32);
    if (lane == 0) atomicAdd(&red[t], v);
  }
  __syncthreads();
  if (tid < TN) {
    float z = 0.0f;
    for (int t = 0; t < TN; ++t) z += (red[t] * (1.0f / HW)) * se_w1[t];
    z = fmaxf(z, 0.0f);
    sse[tid] = sigmoidf_(z * se_w2[tid]);
  }
  __syncthreads();

  const float rinv_bin  = 1.0f / (softplusf_(bin_size_param[0]) + 1e-6f);
  const float rinv_temp = 1.0f / (softplusf_(soft_temp_param[0]) + 1e-6f);

  for (int pp = 0; pp < 2; ++pp) {
    int p = tid + pp * 512;
    v8h r0 = *(const v8h*)(piece1 + p * 16);
    v8h r1 = *(const v8h*)(piece1 + p * 16 + 8);
    v2h r2 = *(const v2h*)(piece2 + p * 8);
    _Float16 h16[TN];
#pragma unroll
    for (int t = 0; t < TN; ++t) {
      float f = (float)((t < 8) ? r0[t & 7] : (t < 16) ? r1[t & 7] : r2[t & 1]) * sse[t];
      float q = f * rinv_bin;
      float d[NB]; float dmin = 1e30f;
#pragma unroll
      for (int k = 0; k < NB; ++k) {
        float dd = q - bc[k]; dd *= dd; d[k] = dd; dmin = fminf(dmin, dd);
      }
      float num = 0.0f, den = 0.0f;
#pragma unroll
      for (int k = 0; k < NB; ++k) {
        float w = __expf(-(d[k] - dmin) * rinv_temp);
        num += w * bc[k]; den += w;
      }
      h16[t] = (_Float16)(num / den);
    }
    v8h s0, s1;
#pragma unroll
    for (int j = 0; j < 8; ++j) { s0[j] = h16[j]; s1[j] = h16[8 + j]; }
    *(v8h*)(piece1 + p * 16)     = s0;
    *(v8h*)(piece1 + p * 16 + 8) = s1;
    v2h s2; s2[0] = h16[16]; s2[1] = h16[17];
    *(v2h*)(piece2 + p * 8) = s2;
  }
  __syncthreads();

  // --- WMMA: soft_q [1024 x 32] @ agg_w1^T [32 x 64] + relu + dot(w2) ---
  const int lnib = lane & 15;
  const int hig  = (lane >= 16) ? 8 : 0;   // second 8-half group selector

  // B fragments for the 4 N-tiles: two unconditional b128 loads each
  v16h bfr[4];
#pragma unroll
  for (int nt = 0; nt < 4; ++nt) {
    int n = nt * 16 + lnib;
    v8h b0 = *(const v8h*)(wB + n * KPAD + hig);        // K hig..hig+7
    v8h b1 = *(const v8h*)(wB + n * KPAD + 16 + hig);   // K 16+hig..23+hig
    bfr[nt] = __builtin_shufflevector(b0, b1,
        0, 1, 2, 3, 4, 5, 6, 7, 8, 9, 10, 11, 12, 13, 14, 15);
  }
  const float b2v = agg_b2[0];
  const int p2row_zero = HW;               // all-zero row of piece2

  for (int i = 0; i < 4; ++i) {
    int mt = wav * 4 + i;                  // 16 waves x 4 = 64 M-tiles
    int m  = mt * 16 + lnib;
    // chunk1: K hig..hig+7 from piece1 (one b128)
    v8h a0 = *(const v8h*)(piece1 + m * 16 + hig);
    // chunk2: lanes<16 -> K16..23 (piece2 row m); lanes>=16 -> K24..31 = zeros
    int p2r = (lane < 16) ? m : p2row_zero;              // branchless cndmask
    v8h a1 = *(const v8h*)(piece2 + p2r * 8);
    v16h afr = __builtin_shufflevector(a0, a1,
        0, 1, 2, 3, 4, 5, 6, 7, 8, 9, 10, 11, 12, 13, 14, 15);

    float acc8[8];
#pragma unroll
    for (int r = 0; r < 8; ++r) acc8[r] = 0.0f;
#pragma unroll
    for (int nt = 0; nt < 4; ++nt) {
      v8f cf = {};
      cf = __builtin_amdgcn_wmma_f32_16x16x32_f16(
          /*neg_a=*/false, afr, /*neg_b=*/false, bfr[nt],
          /*c_mod=*/(short)0, cf, /*reuse_a=*/false, /*reuse_b=*/false);
#pragma unroll
      for (int r = 0; r < 8; ++r) {
        int n = nt * 16 + lnib;
        float h = cf[r] + b1s[n];
        h = fmaxf(h, 0.0f);
        acc8[r] += h * w2s[n];             // fold agg_w2 into epilogue
      }
    }
    // sum the 64 feature columns: xor-reduce within each 16-lane half
#pragma unroll
    for (int r = 0; r < 8; ++r) {
      float tot = acc8[r];
      tot += __shfl_xor(tot, 1, 32);
      tot += __shfl_xor(tot, 2, 32);
      tot += __shfl_xor(tot, 4, 32);
      tot += __shfl_xor(tot, 8, 32);
      if (lnib == 0) pv[mt * 16 + r + hig] = tot + b2v;
    }
  }
  __syncthreads();

  // GeM pool partial: mean of clip(pv)^p over the image
  const float pgem = gem_p[0];
  float lg = 0.0f;
  for (int pp = 0; pp < 2; ++pp) {
    float v = fmaxf(pv[tid + pp * 512], 1e-6f);
    lg += __powf(v, pgem);
  }
  for (int mk = 1; mk < 32; mk <<= 1) lg += __shfl_xor(lg, mk, 32);
  if (lane == 0) atomicAdd(&gacc, lg);
  __syncthreads();
  if (tid == 0) gpool[idx] = __powf(gacc * (1.0f / HW), 1.0f / pgem);
}

// ---------------------------------------------------------------------------
// Kernel C: Linear(3, 256) + LayerNorm + L2 normalize. One block per batch row.
// ---------------------------------------------------------------------------
__global__ __launch_bounds__(FD) void head_kernel(
    const float* __restrict__ gpool,
    const float* __restrict__ pool_w,
    const float* __restrict__ pool_b,
    const float* __restrict__ ln_w,
    const float* __restrict__ ln_b,
    float* __restrict__ out)
{
  __shared__ float red[FD];
  const int b = blockIdx.x, f = threadIdx.x;
  float g0 = gpool[b * 3 + 0], g1 = gpool[b * 3 + 1], g2 = gpool[b * 3 + 2];
  float o = pool_b[f] + g0 * pool_w[f * 3 + 0] + g1 * pool_w[f * 3 + 1] +
            g2 * pool_w[f * 3 + 2];

  red[f] = o; __syncthreads();
  for (int s = FD / 2; s > 0; s >>= 1) { if (f < s) red[f] += red[f + s]; __syncthreads(); }
  float mean = red[0] * (1.0f / FD); __syncthreads();

  float dv = o - mean;
  red[f] = dv * dv; __syncthreads();
  for (int s = FD / 2; s > 0; s >>= 1) { if (f < s) red[f] += red[f + s]; __syncthreads(); }
  float var = red[0] * (1.0f / FD); __syncthreads();

  float on = dv * rsqrtf(var + 1e-5f) * ln_w[f] + ln_b[f];
  red[f] = on * on; __syncthreads();
  for (int s = FD / 2; s > 0; s >>= 1) { if (f < s) red[f] += red[f + s]; __syncthreads(); }
  float nrm = sqrtf(red[0]);
  out[b * FD + f] = on / fmaxf(nrm, 1e-12f);
}

// ---------------------------------------------------------------------------
extern "C" void kernel_launch(void* const* d_in, const int* in_sizes, int n_in,
                              void* d_out, int out_size, void* d_ws, size_t ws_size,
                              hipStream_t stream)
{
  const float* x            = (const float*)d_in[0];
  const float* radius_param = (const float*)d_in[1];
  const float* bin_size_p   = (const float*)d_in[2];
  const float* soft_temp_p  = (const float*)d_in[3];
  const float* bin_centers  = (const float*)d_in[4];
  const float* offset_w     = (const float*)d_in[5];
  const float* offset_b     = (const float*)d_in[6];
  const float* norm_w       = (const float*)d_in[7];
  const float* norm_b       = (const float*)d_in[8];
  const float* se_w1        = (const float*)d_in[9];
  const float* se_w2        = (const float*)d_in[10];
  const float* agg_w1       = (const float*)d_in[11];
  const float* agg_b1       = (const float*)d_in[12];
  const float* agg_w2       = (const float*)d_in[13];
  const float* agg_b2       = (const float*)d_in[14];
  const float* gem_p        = (const float*)d_in[15];
  const float* pool_w       = (const float*)d_in[16];
  const float* pool_b       = (const float*)d_in[17];
  const float* ln_w         = (const float*)d_in[18];
  const float* ln_b         = (const float*)d_in[19];
  float* out = (float*)d_out;

  float* gmap  = (float*)d_ws;          // 768*1024 floats (3 MB, lives in L2)
  float* gpool = gmap + 768 * 1024;     // 768 floats

  gradnorm_kernel<<<768, 256, 0, stream>>>(x, norm_w, norm_b, gmap);
  lpmp_main_kernel<<<768, 512, 0, stream>>>(gmap, radius_param, bin_size_p,
      soft_temp_p, bin_centers, offset_w, offset_b, se_w1, se_w2,
      agg_w1, agg_b1, agg_w2, agg_b2, gem_p, gpool);
  head_kernel<<<256, FD, 0, stream>>>(gpool, pool_w, pool_b, ln_w, ln_b, out);
}
